// RtStgcnLayer_26929444946465
// MI455X (gfx1250) — compile-verified, matrix operand order checked by reading, not run on hardware
//
#include <hip/hip_runtime.h>

typedef __attribute__((ext_vector_type(16))) __bf16       v16bf;
typedef __attribute__((ext_vector_type(8)))  float        v8f;
typedef __attribute__((ext_vector_type(4)))  unsigned int v4u;
typedef __attribute__((ext_vector_type(8)))  int          v8i;
typedef __attribute__((ext_vector_type(4)))  int          v4i;

#define N_BATCH 16
#define C_IN    64
#define L_DIM   2048
#define V_DIM   25
#define P_DIM   3
#define O_DIM   192       // 3 * 64
#define LB      16        // output l positions per workgroup
#define HALO    8         // KERNEL_SIZE - 1
#define LT      (LB + HALO)
#define CH      2         // l values per chunk
#define NCHUNK  (LT / CH)
#define KSZ     9
#define ROW_STRIDE (L_DIM * V_DIM)   // 51200 floats between c rows of x

// ---- fp32 -> bf16 (round to nearest even) ----
__device__ __forceinline__ unsigned short f2bf(float f) {
  unsigned int u = __builtin_bit_cast(unsigned int, f);
  u += 0x7FFFu + ((u >> 16) & 1u);
  return (unsigned short)(u >> 16);
}

// Fragment-major LDS layout: each panel = one 16(MN)x32(K) bf16 operand tile,
// 512 elements; lane's 16 elements are contiguous (32B -> 2x ds_load_b128).
// Slot of element (m_or_n, k') inside a panel, per CDNA5 ISA 7.12.2:
//   lanes 0-15:  M=lane,    VGPR0-3 K=0..7,  VGPR4-7 K=16..23
//   lanes 16-31: M=lane-16, VGPR0-3 K=8..15, VGPR4-7 K=24..31
__device__ __forceinline__ int frag_slot(int mn, int kp) {
  int hi = (kp >> 3) & 1;                    // which lane half holds this K
  int j  = (kp & 7) + ((kp >= 16) ? 8 : 0);  // element index within lane
  return (hi * 16 + mn) * 16 + j;
}

// ---- Tensor Data Mover: load one (64 x 50 fp32) x-chunk into LDS staging ----
// D# group0: count=1 | lds_addr | global_addr | type=2
// D# group1: data_size=4B, tensor_dim0=51200, tensor_dim1=64,
//            tile_dim0=50, tile_dim1=64, tensor_dim0_stride=51200
__device__ __forceinline__ void tdm_issue_x(const float* gptr, unsigned lds_off) {
  unsigned long long ga = (unsigned long long)(uintptr_t)gptr;
  v4u g0;
  g0[0] = 1u;                                    // count=1 (valid user D#)
  g0[1] = lds_off;                               // lds_addr (bytes)
  g0[2] = (unsigned)(ga & 0xFFFFFFFFull);        // global_addr[31:0]
  g0[3] = (unsigned)(ga >> 32) | 0x80000000u;    // global_addr[56:32] | type=2
  v8i g1;
  g1[0] = 0x20000;                               // data_size=2 -> 4 bytes
  g1[1] = (int)((ROW_STRIDE & 0xFFFFu) << 16);   // tensor_dim0 lo16
  g1[2] = (int)((ROW_STRIDE >> 16) | (C_IN << 16)); // td0 hi16 | tensor_dim1 lo16
  g1[3] = (int)((CH * V_DIM) << 16);             // td1 hi16 | tile_dim0 = 50
  g1[4] = C_IN;                                  // tile_dim1=64, tile_dim2=0
  g1[5] = ROW_STRIDE;                            // tensor_dim0_stride lo32
  g1[6] = 0;
  g1[7] = 0;
  v4i z4 = (v4i)0;                               // D# groups 2/3 unused (2-D tensor)
  v8i z8 = (v8i)0;
  __builtin_amdgcn_tensor_load_to_lds(g0, g1, z4, z4, z8, 0);
}

__global__ __launch_bounds__(256) void rt_stgcn_fused(
    const float* __restrict__ x,   // (16,64,2048,25)
    const float* __restrict__ A,   // (3,25,25)
    const float* __restrict__ Wc,  // (192,64)
    const float* __restrict__ bc,  // (192,)
    float* __restrict__ out) {     // (16,64,2048,25)
  __shared__ unsigned short W_fr[24 * 512];            // (rt 0..11, kt 0..1) panels
  __shared__ unsigned short A_fr[6 * 512];             // (p 0..2, nt 0..1) panels
  __shared__ unsigned short a_fr[24 * 512];            // (rt 0..11, lc 0..1) panels
  __shared__ unsigned short x_fr[8 * 512];             // (nt 0..3, kt 0..1) panels
  __shared__ float          b_lds[O_DIM];
  __shared__ float          stage[2][C_IN * CH * V_DIM]; // TDM double buffer (fp32)

  const int tid  = threadIdx.x;
  const int lane = tid & 31;
  const int wid  = tid >> 5;
  const int nb   = blockIdx.x / (L_DIM / LB);
  const int l0   = (blockIdx.x % (L_DIM / LB)) * LB;

  const float* xb = x   + (size_t)nb * C_IN * L_DIM * V_DIM;
  float*       ob = out + (size_t)nb * C_IN * L_DIM * V_DIM;

  // ---- kick off TDM for chunk 0 (skip if fully in causal zero-pad) ----
  if (wid == 0 && (l0 - HALO) >= 0)
    tdm_issue_x(xb + (size_t)(l0 - HALO) * V_DIM,
                (unsigned)(uintptr_t)&stage[0][0]);

  // ---- one-time weight staging into fragment-major panels ----
  for (int i = tid; i < O_DIM * C_IN; i += 256) {      // conv_w (192x64)
    int o = i >> 6, ci = i & 63;
    int idx = ((o >> 4) * 2 + (ci >> 5)) * 512 + frag_slot(o & 15, ci & 31);
    W_fr[idx] = f2bf(Wc[i]);
  }
  for (int i = tid; i < O_DIM; i += 256) b_lds[i] = bc[i];
  for (int i = tid; i < P_DIM * 1024; i += 256) {      // A padded to 3x32x32
    int p = i >> 10, v = (i >> 5) & 31, w = i & 31;
    float val = (v < V_DIM && w < V_DIM) ? A[p * (V_DIM * V_DIM) + v * V_DIM + w] : 0.f;
    int idx = (p * 2 + (w >> 4)) * 512 + frag_slot(w & 15, v);  // B operand: N=w, K=v
    A_fr[idx] = f2bf(val);
  }
  __syncthreads();

  // each wave owns one 16x16 tile of es(64x32): rows c, cols w
  const int mi2 = wid >> 1;   // 0..3  -> c block
  const int ni2 = wid & 1;    // 0..1  -> w block
  const v16bf* Wv = (const v16bf*)W_fr;
  const v16bf* Av = (const v16bf*)A_fr;
  const v16bf* av = (const v16bf*)a_fr;
  const v16bf* xv = (const v16bf*)x_fr;

  v16bf bA[P_DIM];            // A_p B-fragments live in registers all kernel
#pragma unroll
  for (int p = 0; p < P_DIM; ++p) bA[p] = Av[(p * 2 + ni2) * 32 + lane];

  const v8f vzero = {0.f, 0.f, 0.f, 0.f, 0.f, 0.f, 0.f, 0.f};
  v8f ring[KSZ];              // sliding 9-tap window of es tiles (registers)
#pragma unroll
  for (int q = 0; q < KSZ; ++q) ring[q] = vzero;

  for (int ch = 0; ch < NCHUNK; ++ch) {
    const int lbase = l0 - HALO + ch * CH;
    const int lnext = lbase + CH;
    const bool issue_next = (ch + 1 < NCHUNK) && (lnext >= 0);

    // issue next chunk's DMA, then ensure current chunk's DMA has landed
    if (wid == 0) {
      if (issue_next) {
        tdm_issue_x(xb + (size_t)lnext * V_DIM,
                    (unsigned)(uintptr_t)&stage[(ch + 1) & 1][0]);
        __builtin_amdgcn_s_wait_tensorcnt(1);   // oldest (current chunk) done
      } else {
        __builtin_amdgcn_s_wait_tensorcnt(0);
      }
    }
    __syncthreads();

    // ---- convert staged fp32 -> fragment-major bf16 (zero for l<0, v>=25) ----
    const float* stg = stage[ch & 1];
    for (int i = tid; i < C_IN * 64; i += 256) {
      int ci = i >> 6, col = i & 63;
      int lc = col >> 5, v = col & 31;
      float val = (lbase >= 0 && v < V_DIM) ? stg[ci * (CH * V_DIM) + lc * V_DIM + v] : 0.f;
      int idx = ((col >> 4) * 2 + (ci >> 5)) * 512 + frag_slot(col & 15, ci & 31);
      x_fr[idx] = f2bf(val);
    }
    __syncthreads();

    // ---- stage 1: a = W(192x64) @ x(64x64cols) + bias, bf16 WMMA, fp32 acc ----
#pragma unroll
    for (int i = 0; i < 6; ++i) {
      int t  = wid + 8 * i;            // 48 tiles over 8 waves
      int mi = t >> 2, ni = t & 3;
      v8f acc = vzero;
#pragma unroll
      for (int k = 0; k < 2; ++k) {
        v16bf fa = Wv[(mi * 2 + k) * 32 + lane];   // 2x ds_load_b128
        v16bf fb = xv[(ni * 2 + k) * 32 + lane];
        acc = __builtin_amdgcn_wmma_f32_16x16x32_bf16(
            false, fa, false, fb, (short)0, acc, false, false);
      }
      int n_   = lane & 15;
      int kprm = (ni & 1) * 16 + n_;   // column within the l-slice (v index)
      int lc_  = ni >> 1;
      int rofs = (lane & 16) ? 8 : 0;
      int hi   = (kprm >> 3) & 1;
      int jel  = (kprm & 7) + ((kprm >= 16) ? 8 : 0);
#pragma unroll
      for (int r = 0; r < 8; ++r) {
        int m = r + rofs;
        int o = mi * 16 + m;
        a_fr[(mi * 2 + lc_) * 512 + (hi * 16 + m) * 16 + jel] = f2bf(acc[r] + b_lds[o]);
      }
    }
    __syncthreads();

    // ---- stage 2: es_l = sum_p a_p(64x32) @ A_p(32x32); 9-tap causal sum ----
#pragma unroll
    for (int lc = 0; lc < CH; ++lc) {
      const int l = lbase + lc;        // block-uniform -> EXEC stays full
      v8f es = vzero;
      if (l >= 0) {
#pragma unroll
        for (int p = 0; p < P_DIM; ++p) {
          v16bf fa = av[((p * 4 + mi2) * 2 + lc) * 32 + lane];
          es = __builtin_amdgcn_wmma_f32_16x16x32_bf16(
              false, fa, false, bA[p], (short)0, es, false, false);
        }
      }
#pragma unroll
      for (int q = 0; q < KSZ - 1; ++q) ring[q] = ring[q + 1];
      ring[KSZ - 1] = es;

      if (l >= l0) {
        v8f h = ring[0];
#pragma unroll
        for (int q = 1; q < KSZ; ++q) h = h + ring[q];
        int w    = ni2 * 16 + (lane & 15);
        int rofs = (lane & 16) ? 8 : 0;
        if (w < V_DIM) {
#pragma unroll
          for (int r = 0; r < 8; ++r) {
            int c = mi2 * 16 + r + rofs;
            size_t idx = ((size_t)c * L_DIM + l) * V_DIM + w;
            float val = h[r] + xb[idx];       // residual
            ob[idx] = val > 0.f ? val : 0.f;  // relu
          }
        }
      }
    }
    __syncthreads();
  }
}

extern "C" void kernel_launch(void* const* d_in, const int* in_sizes, int n_in,
                              void* d_out, int out_size, void* d_ws, size_t ws_size,
                              hipStream_t stream) {
  (void)in_sizes; (void)n_in; (void)out_size; (void)d_ws; (void)ws_size;
  const float* x  = (const float*)d_in[0];
  const float* A  = (const float*)d_in[1];
  const float* Wc = (const float*)d_in[2];
  const float* bc = (const float*)d_in[3];
  float* out = (float*)d_out;

  dim3 grid(N_BATCH * (L_DIM / LB));  // 16 * 128 = 2048 workgroups
  rt_stgcn_fused<<<grid, 256, 0, stream>>>(x, A, Wc, bc, out);
}
